// Attention_42202348651096
// MI455X (gfx1250) — matrix operand order repaired and checked by reading, compiled, not run on hardware
//
#include <hip/hip_runtime.h>
#include <hip/hip_bf16.h>

#define SEQ   4096
#define DM    1024
#define NHEAD 16
#define DK    64

typedef __bf16 bf16_t;
typedef __attribute__((ext_vector_type(8)))  __bf16 bf16x8;
typedef __attribute__((ext_vector_type(16))) __bf16 v16bf;
typedef __attribute__((ext_vector_type(8)))  float  v8f;
typedef int i32x4 __attribute__((vector_size(16)));  // matches builtin pointee

// ---------------------------------------------------------------------------
// WMMA fragment helpers (wave32, 16x16x32 bf16, contiguous k order, same in A and B)
// ---------------------------------------------------------------------------
static __device__ inline v16bf frag_cat(bf16x8 x0, bf16x8 x1) {
  return __builtin_shufflevector(x0, x1, 0,1,2,3,4,5,6,7,8,9,10,11,12,13,14,15);
}

// A fragment 16(M)x32(K): lane holds row M = row0 + (lane&15).
// elem e -> K = kb + (lane>>4)*16 + e  (two contiguous 16B loads)
static __device__ inline v16bf load_frag_a(const bf16_t* p, int ld, int row0,
                                           int kb, int lane) {
  const bf16_t* r = p + (size_t)(row0 + (lane & 15)) * ld + kb + ((lane >> 4) << 4);
  bf16x8 x0 = *(const bf16x8*)(r);
  bf16x8 x1 = *(const bf16x8*)(r + 8);
  return frag_cat(x0, x1);
}

// B fragment 32(K)x16(N) loaded from K-contiguous storage Bt[n][k]:
// lane holds col N = n0 + (lane&15); elem e -> K = kb + (lane>>4)*16 + e
static __device__ inline v16bf load_frag_b(const bf16_t* bt, int ld, int n0,
                                           int kb, int lane) {
  const bf16_t* r = bt + (size_t)(n0 + (lane & 15)) * ld + kb + ((lane >> 4) << 4);
  bf16x8 x0 = *(const bf16x8*)(r);
  bf16x8 x1 = *(const bf16x8*)(r + 8);
  return frag_cat(x0, x1);
}

static __device__ inline v8f wmma_bf16(v16bf a, v16bf b, v8f c) {
  return __builtin_amdgcn_wmma_f32_16x16x32_bf16(false, a, false, b, (short)0, c,
                                                 false, false);
}

static __device__ inline v8f vzero8() { v8f z = {0.f,0.f,0.f,0.f,0.f,0.f,0.f,0.f}; return z; }

// ---------------------------------------------------------------------------
// Async global -> LDS staging (CDNA5 GLOBAL_LOAD_ASYNC_TO_LDS_B128, ASYNCcnt).
// Builtin signature (probed from the round-3 diagnostic): param 1 is
// `i32x4 __device__*` (global AS1, source-first), so param 2 is the LDS dest.
// Falls back to a synchronous VGPR round-trip if the builtin is absent.
// ---------------------------------------------------------------------------
#if __has_builtin(__builtin_amdgcn_global_load_async_to_lds_b128)
#define ASYNC_STAGE 1
#else
#define ASYNC_STAGE 0
#endif

static __device__ inline void stage16(const bf16_t* g, bf16_t* l) {
#if ASYNC_STAGE
  __builtin_amdgcn_global_load_async_to_lds_b128(
      (__attribute__((address_space(1))) i32x4*)(i32x4*)g,
      (__attribute__((address_space(3))) i32x4*)(i32x4*)l,
      0, 0);
#else
  *(bf16x8*)l = *(const bf16x8*)g;
#endif
}

static __device__ inline void wait_stage() {
#if ASYNC_STAGE
#if __has_builtin(__builtin_amdgcn_s_wait_asynccnt)
  __builtin_amdgcn_s_wait_asynccnt(0);
#else
  asm volatile("s_wait_asynccnt 0" ::: "memory");
#endif
#endif
}

// ---------------------------------------------------------------------------
// Stage 1: precision conversion + weight transposes (bandwidth trivial)
// ---------------------------------------------------------------------------
__global__ void k_cvt(const float* __restrict__ in, bf16_t* __restrict__ out, int n) {
  int i = blockIdx.x * blockDim.x + threadIdx.x;
  if (i < n) out[i] = (bf16_t)in[i];
}

// w: [batch][rows][cols] (f32) -> wt: [batch][cols][rows] (bf16)
__global__ void k_transpose_w(const float* __restrict__ w, bf16_t* __restrict__ wt,
                              int rows, int cols, int batch) {
  int i = blockIdx.x * blockDim.x + threadIdx.x;
  int total = batch * rows * cols;
  if (i >= total) return;
  int b = i / (rows * cols);
  int rem = i - b * rows * cols;
  int r = rem / cols, c = rem - r * cols;
  wt[(size_t)b * rows * cols + (size_t)c * rows + r] = (bf16_t)w[i];
}

// ---------------------------------------------------------------------------
// Stage 2: per-head projections  q/k = X @ W  (row-major out),  v stored as v^T
// grid: (SEQ/128, NHEAD, 3), block 128 (4 waves); wave owns a 32x64 tile:
// 8 WMMAs per 12 B128 loads per K-step.
// ---------------------------------------------------------------------------
__global__ void k_proj(const bf16_t* __restrict__ Qb, const bf16_t* __restrict__ Kb,
                       const bf16_t* __restrict__ Vb,
                       const bf16_t* __restrict__ WQt, const bf16_t* __restrict__ WKt,
                       const bf16_t* __restrict__ WVt,
                       bf16_t* __restrict__ qh, bf16_t* __restrict__ kh,
                       bf16_t* __restrict__ vt) {
  const int lane = threadIdx.x & 31;
  const int wv   = threadIdx.x >> 5;
  const int h    = blockIdx.y;
  const int which = blockIdx.z;
  const int r0 = blockIdx.x * 128 + wv * 32;

  const bf16_t* X  = (which == 0) ? Qb : (which == 1) ? Kb : Vb;
  const bf16_t* Wt = ((which == 0) ? WQt : (which == 1) ? WKt : WVt) + (size_t)h * DK * DM;

  v8f acc[2][4];
#pragma unroll
  for (int m = 0; m < 2; m++)
#pragma unroll
    for (int nt = 0; nt < 4; nt++) acc[m][nt] = vzero8();

  for (int kb = 0; kb < DM; kb += 32) {
    v16bf a0 = load_frag_a(X, DM, r0,      kb, lane);
    v16bf a1 = load_frag_a(X, DM, r0 + 16, kb, lane);
    v16bf b0 = load_frag_b(Wt, DM,  0, kb, lane);
    v16bf b1 = load_frag_b(Wt, DM, 16, kb, lane);
    v16bf b2 = load_frag_b(Wt, DM, 32, kb, lane);
    v16bf b3 = load_frag_b(Wt, DM, 48, kb, lane);
    acc[0][0] = wmma_bf16(a0, b0, acc[0][0]);
    acc[0][1] = wmma_bf16(a0, b1, acc[0][1]);
    acc[0][2] = wmma_bf16(a0, b2, acc[0][2]);
    acc[0][3] = wmma_bf16(a0, b3, acc[0][3]);
    acc[1][0] = wmma_bf16(a1, b0, acc[1][0]);
    acc[1][1] = wmma_bf16(a1, b1, acc[1][1]);
    acc[1][2] = wmma_bf16(a1, b2, acc[1][2]);
    acc[1][3] = wmma_bf16(a1, b3, acc[1][3]);
  }

  const int hi = lane >> 4, col = lane & 15;
  if (which < 2) {
    bf16_t* out = ((which == 0) ? qh : kh) + (size_t)h * SEQ * DK;
#pragma unroll
    for (int m = 0; m < 2; m++)
#pragma unroll
      for (int nt = 0; nt < 4; nt++)
#pragma unroll
        for (int i = 0; i < 8; i++)
          out[(size_t)(r0 + m * 16 + i + 8 * hi) * DK + nt * 16 + col] =
              (bf16_t)acc[m][nt][i];
  } else {
    bf16_t* out = vt + (size_t)h * DK * SEQ;  // v^T : [dk][seq]
#pragma unroll
    for (int m = 0; m < 2; m++)
#pragma unroll
      for (int nt = 0; nt < 4; nt++) {
        bf16x8 t;
#pragma unroll
        for (int i = 0; i < 8; i++) t[i] = (bf16_t)acc[m][nt][i];
        *(bf16x8*)(out + (size_t)(nt * 16 + col) * SEQ + r0 + m * 16 + 8 * hi) = t;
      }
  }
}

// ---------------------------------------------------------------------------
// Stage 3: flash attention. grid (SEQ/128, NHEAD), block 256 (8 waves).
// Wave owns 16 queries. The block stages each 32-key K/V^T chunk into LDS
// once, double-buffered, via async global->LDS loads (overlapped with the
// current chunk's WMMAs; synced by s_wait_asynccnt + barrier).
// ---------------------------------------------------------------------------
__global__ void k_attn(const bf16_t* __restrict__ qh, const bf16_t* __restrict__ kh,
                       const bf16_t* __restrict__ vt, bf16_t* __restrict__ concat) {
  __shared__ __align__(16) bf16_t kLds[2][32 * DK];   // [buf][key 0..31][dk 0..63]
  __shared__ __align__(16) bf16_t vLds[2][DK * 32];   // [buf][dk 0..63][key 0..31]
  __shared__ __align__(16) bf16_t Pbuf[8][16 * 32];
  __shared__ float Srow[8][16];

  const int tid  = threadIdx.x;
  const int lane = tid & 31;
  const int wv   = tid >> 5;
  const int h    = blockIdx.y;
  const int q0   = (blockIdx.x * 8 + wv) * 16;
  const int hi = lane >> 4, col = lane & 15;
  const float scale = 0.125f;  // 1/sqrt(64)

  const bf16_t* qp = qh + (size_t)h * SEQ * DK;
  const bf16_t* kp = kh + (size_t)h * SEQ * DK;
  const bf16_t* vp = vt + (size_t)h * DK * SEQ;

  // cooperative-staging index precompute (one b128 per thread per buffer)
  const int krow = tid >> 3, kcol = (tid & 7) << 3;    // kLds[.][krow][kcol..+7]
  const int vrow = tid >> 2, vcol = (tid & 3) << 3;    // vLds[.][vrow][vcol..+7]

  // q fragments for this wave's 16 rows, kept resident (K = 0..31, 32..63)
  v16bf aq0 = load_frag_a(qp, DK, q0, 0, lane);
  v16bf aq1 = load_frag_a(qp, DK, q0, 32, lane);

  v8f o0 = vzero8(), o1 = vzero8(), o2 = vzero8(), o3 = vzero8();
  float mi[8], li[8];
#pragma unroll
  for (int i = 0; i < 8; i++) { mi[i] = -1e30f; li[i] = 0.f; }

  // prologue: stage chunk 0 into buffer 0
  stage16(kp + (size_t)krow * DK + kcol, &kLds[0][krow * DK + kcol]);
  stage16(vp + (size_t)vrow * SEQ + vcol, &vLds[0][vrow * 32 + vcol]);

  for (int j = 0; j < SEQ; j += 32) {
    const int cur = (j >> 5) & 1;
    // current buffer's async writes done (this wave), then block-wide sync.
    wait_stage();
    __syncthreads();
    // overlap: stage next chunk into the other buffer while we compute.
    if (j + 32 < SEQ) {
      const int nxt = cur ^ 1;
      stage16(kp + (size_t)(j + 32 + krow) * DK + kcol,
              &kLds[nxt][krow * DK + kcol]);
      stage16(vp + (size_t)vrow * SEQ + j + 32 + vcol,
              &vLds[nxt][vrow * 32 + vcol]);
    }

    // ---- scores for keys [j, j+32) : S = q @ k^T (frags from LDS) ----
    v8f s0 = vzero8(), s1 = vzero8();
    v16bf bk00 = load_frag_b(kLds[cur], DK,  0,  0, lane);
    v16bf bk01 = load_frag_b(kLds[cur], DK,  0, 32, lane);
    v16bf bk10 = load_frag_b(kLds[cur], DK, 16,  0, lane);
    v16bf bk11 = load_frag_b(kLds[cur], DK, 16, 32, lane);
    s0 = wmma_bf16(aq0, bk00, s0);
    s0 = wmma_bf16(aq1, bk01, s0);
    s1 = wmma_bf16(aq0, bk10, s1);
    s1 = wmma_bf16(aq1, bk11, s1);

    // ---- online softmax (rows of S = queries, row i+8*hi across 16 lanes) ----
#pragma unroll
    for (int i = 0; i < 8; i++) {
      float a0 = s0[i] * scale, a1 = s1[i] * scale;
      float rm = fmaxf(a0, a1);
#pragma unroll
      for (int m = 1; m < 16; m <<= 1) rm = fmaxf(rm, __shfl_xor(rm, m, 32));
      float mn = fmaxf(mi[i], rm);
      float al = __expf(mi[i] - mn);
      float p0 = __expf(a0 - mn), p1 = __expf(a1 - mn);
      float rs = p0 + p1;
#pragma unroll
      for (int m = 1; m < 16; m <<= 1) rs += __shfl_xor(rs, m, 32);
      li[i] = li[i] * al + rs;
      mi[i] = mn;
      // P tile -> LDS (row-major [query][key0..31], bf16), wave-private
      Pbuf[wv][(i + 8 * hi) * 32 + col]      = (bf16_t)p0;
      Pbuf[wv][(i + 8 * hi) * 32 + 16 + col] = (bf16_t)p1;
      if (col == 0) Srow[wv][hi * 8 + i] = al;  // per-query rescale factor
    }
    asm volatile("s_wait_dscnt 0" ::: "memory");  // DS in-order per wave

    // rescale running O^T (columns = queries = lane%16)
    float alq = Srow[wv][col];
    o0 *= alq; o1 *= alq; o2 *= alq; o3 *= alq;

    // B = P^T from LDS: lane reads P[query=col][keys (lane/16)*16 + 0..15]
    const bf16x8* pp = (const bf16x8*)&Pbuf[wv][col * 32 + hi * 16];
    v16bf bP = frag_cat(pp[0], pp[1]);

    // O^T += v^T(chunk) @ P^T  (A rows = dk, contiguous loads from LDS v^T)
    v16bf av0 = load_frag_a(vLds[cur], 32,  0, 0, lane);
    v16bf av1 = load_frag_a(vLds[cur], 32, 16, 0, lane);
    v16bf av2 = load_frag_a(vLds[cur], 32, 32, 0, lane);
    v16bf av3 = load_frag_a(vLds[cur], 32, 48, 0, lane);
    o0 = wmma_bf16(av0, bP, o0);
    o1 = wmma_bf16(av1, bP, o1);
    o2 = wmma_bf16(av2, bP, o2);
    o3 = wmma_bf16(av3, bP, o3);
  }

  // final 1/l per query (bounce rows -> per-lane scalar)
  if (col == 0) {
#pragma unroll
    for (int i = 0; i < 8; i++) Srow[wv][hi * 8 + i] = li[i];
  }
  asm volatile("s_wait_dscnt 0" ::: "memory");
  float lrec = 1.0f / Srow[wv][col];

  // O^T tile t: rows dk = t*16 + i + 8*hi, col query = col. Store to concat
  // [seq][1024] at column h*64+dk (contiguous over i -> one 16B store each).
  bf16_t* crow = concat + (size_t)(q0 + col) * DM + h * DK;
  v8f os[4] = {o0, o1, o2, o3};
#pragma unroll
  for (int t = 0; t < 4; t++) {
    bf16x8 tv;
#pragma unroll
    for (int i = 0; i < 8; i++) tv[i] = (bf16_t)(os[t][i] * lrec);
    *(bf16x8*)(crow + t * 16 + 8 * hi) = tv;
  }
}

// ---------------------------------------------------------------------------
// Stage 4: out = concat @ W_O  (f32 out). grid (SEQ/128, DM/64), block 128
// (4 waves); wave owns a 32x64 tile: 8 WMMAs per 12 B128 loads per K-step.
// ---------------------------------------------------------------------------
__global__ void k_ogemm(const bf16_t* __restrict__ Cc, const bf16_t* __restrict__ WOt,
                        float* __restrict__ out) {
  const int lane = threadIdx.x & 31;
  const int wv   = threadIdx.x >> 5;
  const int r0 = blockIdx.x * 128 + wv * 32;
  const int n0 = blockIdx.y * 64;

  v8f acc[2][4];
#pragma unroll
  for (int m = 0; m < 2; m++)
#pragma unroll
    for (int nt = 0; nt < 4; nt++) acc[m][nt] = vzero8();

  for (int kb = 0; kb < DM; kb += 32) {
    v16bf a0 = load_frag_a(Cc, DM, r0,      kb, lane);
    v16bf a1 = load_frag_a(Cc, DM, r0 + 16, kb, lane);
    v16bf b0 = load_frag_b(WOt, DM, n0,      kb, lane);
    v16bf b1 = load_frag_b(WOt, DM, n0 + 16, kb, lane);
    v16bf b2 = load_frag_b(WOt, DM, n0 + 32, kb, lane);
    v16bf b3 = load_frag_b(WOt, DM, n0 + 48, kb, lane);
    acc[0][0] = wmma_bf16(a0, b0, acc[0][0]);
    acc[0][1] = wmma_bf16(a0, b1, acc[0][1]);
    acc[0][2] = wmma_bf16(a0, b2, acc[0][2]);
    acc[0][3] = wmma_bf16(a0, b3, acc[0][3]);
    acc[1][0] = wmma_bf16(a1, b0, acc[1][0]);
    acc[1][1] = wmma_bf16(a1, b1, acc[1][1]);
    acc[1][2] = wmma_bf16(a1, b2, acc[1][2]);
    acc[1][3] = wmma_bf16(a1, b3, acc[1][3]);
  }
  const int hi = lane >> 4, col = lane & 15;
#pragma unroll
  for (int m = 0; m < 2; m++)
#pragma unroll
    for (int nt = 0; nt < 4; nt++)
#pragma unroll
      for (int i = 0; i < 8; i++)
        out[(size_t)(r0 + m * 16 + i + 8 * hi) * DM + n0 + nt * 16 + col] =
            acc[m][nt][i];
}

// ---------------------------------------------------------------------------
// Launch. Workspace layout (bf16 elements), total 64 MB.
// ---------------------------------------------------------------------------
extern "C" void kernel_launch(void* const* d_in, const int* in_sizes, int n_in,
                              void* d_out, int out_size, void* d_ws, size_t ws_size,
                              hipStream_t stream) {
  const float* Q  = (const float*)d_in[0];
  const float* K  = (const float*)d_in[1];
  const float* V  = (const float*)d_in[2];
  const float* WQ = (const float*)d_in[3];
  const float* WK = (const float*)d_in[4];
  const float* WV = (const float*)d_in[5];
  const float* WO = (const float*)d_in[6];
  float* out = (float*)d_out;

  bf16_t* ws = (bf16_t*)d_ws;
  const size_t SD = (size_t)SEQ * DM;          // 4,194,304
  const size_t WH = (size_t)NHEAD * DM * DK;   // 1,048,576
  bf16_t* Qb   = ws;
  bf16_t* Kb   = Qb + SD;
  bf16_t* Vb   = Kb + SD;
  bf16_t* WQt  = Vb + SD;
  bf16_t* WKt  = WQt + WH;
  bf16_t* WVt  = WKt + WH;
  bf16_t* WOt  = WVt + WH;
  bf16_t* qhB  = WOt + (size_t)DM * DM;
  bf16_t* khB  = qhB + SD;                     // [h][seq][dk]
  bf16_t* vtB  = khB + SD;                     // [h][dk][seq]
  bf16_t* conc = vtB + SD;                     // [seq][h*dk]

  // Stage 1: conversions / transposes
  {
    int n = (int)SD;
    int nb = (n + 255) / 256;
    k_cvt<<<nb, 256, 0, stream>>>(Q, Qb, n);
    k_cvt<<<nb, 256, 0, stream>>>(K, Kb, n);
    k_cvt<<<nb, 256, 0, stream>>>(V, Vb, n);
    int nw = (int)WH;
    int nbw = (nw + 255) / 256;
    k_transpose_w<<<nbw, 256, 0, stream>>>(WQ, WQt, DM, DK, NHEAD);
    k_transpose_w<<<nbw, 256, 0, stream>>>(WK, WKt, DM, DK, NHEAD);
    k_transpose_w<<<nbw, 256, 0, stream>>>(WV, WVt, DM, DK, NHEAD);
    k_transpose_w<<<nbw, 256, 0, stream>>>(WO, WOt, DM, DM, 1);
  }
  // Stage 2: projections
  k_proj<<<dim3(SEQ / 128, NHEAD, 3), 128, 0, stream>>>(Qb, Kb, Vb, WQt, WKt, WVt,
                                                        qhB, khB, vtB);
  // Stage 3: attention
  k_attn<<<dim3(SEQ / 128, NHEAD), 256, 0, stream>>>(qhB, khB, vtB, conc);
  // Stage 4: output projection
  k_ogemm<<<dim3(SEQ / 128, DM / 64), 128, 0, stream>>>(conc, WOt, out);
}
